// MHA_RoPE_44220983280379
// MI455X (gfx1250) — matrix operand order repaired and checked by reading, not compiled, not run on hardware
//
#include <hip/hip_runtime.h>
#include <hip/hip_bf16.h>
#include <math.h>

typedef __attribute__((ext_vector_type(16))) __bf16 v16bf;
typedef __attribute__((ext_vector_type(8)))  __bf16 v8bf;
typedef __attribute__((ext_vector_type(8)))  float  v8f;

#define B_  4
#define T_  2048
#define D_  1024
#define H_  16
#define HD_ 64
#define M_  (B_*T_)

static __device__ __forceinline__ v8f wmma_bf16(v16bf a, v16bf b, v8f c) {
  // (neg_a, A, neg_b, B, c_mod, C, reuse_a, reuse_b)
  return __builtin_amdgcn_wmma_f32_16x16x32_bf16(false, a, false, b, (short)0, c, false, false);
}

union U16 { v16bf v; v8bf h8[2]; };

// Async DMA: 16B per lane, global -> LDS, tracked by ASYNCcnt.
static __device__ __forceinline__ void async_ld_b128(unsigned lds_off, const void* g) {
  asm volatile("global_load_async_to_lds_b128 %0, %1, off"
               :: "v"(lds_off), "v"((unsigned long long)(uintptr_t)g) : "memory");
}
static __device__ __forceinline__ void wait_async0() {
  asm volatile("s_wait_asynccnt 0x0" ::: "memory");
}
static __device__ __forceinline__ unsigned lds_off32(const void* p) {
  // low 32 bits of a generic LDS pointer = wave-relative LDS byte offset
  return (unsigned)(uintptr_t)p;
}

// Single-instruction lane-xor shuffle: ds_swizzle_b32 group-of-32 mode,
// offset = { xor_mask[14:10], or_mask[9:5]=0, and_mask[4:0]=0x1f }.
template<int XORM>
static __device__ __forceinline__ float swz(float v) {
  return __int_as_float(
      __builtin_amdgcn_ds_swizzle(__float_as_int(v), (XORM << 10) | 0x1f));
}

// ---------------------------------------------------------------------------
// fp32 -> bf16 elementwise convert
// ---------------------------------------------------------------------------
__global__ void __launch_bounds__(256)
cvt_f32_bf16(const float* __restrict__ in, __bf16* __restrict__ out, int n) {
  int i = blockIdx.x * blockDim.x + threadIdx.x;
  int stride = gridDim.x * blockDim.x;
  for (; i < n; i += stride) out[i] = (__bf16)in[i];
}

// ---------------------------------------------------------------------------
// Y = (X(MxK,bf16) @ W(NxK,bf16)^T + bias) * outScale, one wave = 16x64 tile.
// MODE 0: RoPE epilogue, bf16 out in (B,H,T,HD)      (for Q and K)
// MODE 1: bf16 out transposed (B,H,HD,T)             (for V)
// MODE 2: fp32 out row-major (M,N) + bias            (final projection)
// outScale: 1/sqrt(HD) folded into Q (exact: 2^-3); 1.0 elsewhere.
// ---------------------------------------------------------------------------
template<int MODE>
__global__ void __launch_bounds__(128)
gemm_wmma(const __bf16* __restrict__ X, const __bf16* __restrict__ W,
          const float* __restrict__ bias, void* __restrict__ outp,
          int M, int N, int K, float outScale) {
  const int lane = threadIdx.x & 31;
  const int wave = threadIdx.x >> 5;
  const int half = lane >> 4;
  const int col  = lane & 15;
  const int m0 = blockIdx.y * 64 + wave * 16;
  const int n0 = blockIdx.x * 64;

  v8f acc[4] = {};

  for (int kk = 0; kk < K; kk += 32) {
    // A fragment: lane row = m0+col; elems j -> K=kk+half*16+j
    const __bf16* ap = X + (size_t)(m0 + col) * K + kk + half * 16;
    U16 a;
    a.h8[0] = *(const v8bf*)(ap);
    a.h8[1] = *(const v8bf*)(ap + 8);
#pragma unroll
    for (int wn = 0; wn < 4; ++wn) {
      // B fragment: lane col n = n0+wn*16+col ; elems j -> K=kk+half*16+j (row of W)
      const __bf16* bp = W + (size_t)(n0 + wn * 16 + col) * K + kk + half * 16;
      v16bf bf_ = *(const v16bf*)bp;
      acc[wn] = wmma_bf16(a.v, bf_, acc[wn]);
    }
  }

#pragma unroll
  for (int wn = 0; wn < 4; ++wn) {
    const int c = n0 + wn * 16 + col;
    const float bs = bias ? bias[c] : 0.0f;
#pragma unroll
    for (int r = 0; r < 8; ++r) {
      const int m = m0 + half * 8 + r;
      float v = (acc[wn][r] + bs) * outScale;
      if (MODE == 0) {                      // RoPE, out (B,H,T,HD) bf16
        const int b = m >> 11, t = m & (T_ - 1);
        const int h = c >> 6,  i = c & (HD_ - 1);
        const float p = swz<1>(v);                     // partner column c^1
        const float ang = (float)t * __powf(10000.0f, -(float)(i & 31) * (1.0f / 32.0f));
        float sn, cs; __sincosf(ang, &sn, &cs);
        const float res = v * cs + ((i & 1) ? p : -p) * sn;
        ((__bf16*)outp)[(((size_t)(b * H_ + h) * T_ + t) * HD_) + i] = (__bf16)res;
      } else if (MODE == 1) {               // out (B,H,HD,T) bf16 (transposed V)
        const int b = m >> 11, t = m & (T_ - 1);
        const int h = c >> 6,  i = c & (HD_ - 1);
        ((__bf16*)outp)[(((size_t)(b * H_ + h) * HD_ + i) * T_) + t] = (__bf16)v;
      } else {                              // fp32 row-major
        ((float*)outp)[(size_t)m * N + c] = v;
      }
    }
  }
}

// ---------------------------------------------------------------------------
// Flash attention: block = 4 waves x 16 q-rows, 64-key blocks, K/V tiles
// double-buffered in LDS via global_load_async_to_lds_b128 (ASYNCcnt).
// Q pre-scaled by 1/sqrt(HD), softmax runs in place on the S accumulators,
// cross-lane reductions use single-op ds_swizzle stages (8 rows pipelined).
// qr,kr: (B,H,T,HD) bf16. vt: (B,H,HD,T) bf16. ao: (B,T,D) bf16.
// ---------------------------------------------------------------------------
__global__ void __launch_bounds__(128)
flash_attn(const __bf16* __restrict__ qr, const __bf16* __restrict__ kr,
           const __bf16* __restrict__ vt, __bf16* __restrict__ ao) {
  __shared__ __bf16 ktile[2][64][64];   // [buf][key][hd]      8KB each
  __shared__ __bf16 vtile[2][64][64];   // [buf][hd][key]      8KB each
  __shared__ __bf16 plds[4][16][64];    // per-wave P staging  8KB total
  const int tid  = threadIdx.x;
  const int lane = tid & 31;
  const int wave = tid >> 5;
  const int half = lane >> 4;
  const int col  = lane & 15;
  const int bh = blockIdx.y;                 // b*H + h
  const int b = bh >> 4, h = bh & 15;
  const int q0 = blockIdx.x * 64 + wave * 16;

  const __bf16* qb = qr + (size_t)bh * T_ * HD_;
  const __bf16* kb = kr + (size_t)bh * T_ * HD_;
  const __bf16* vb = vt + (size_t)bh * HD_ * T_;

  // cooperative async stage of one 64-key K tile + V tile into buffer `buf`
  auto stage = [&](int buf, int kblk) {
    // K tile: 64 rows x 64 hd bf16 = 8KB, contiguous in global
    const __bf16* kg = kb + (size_t)kblk * HD_;
    const unsigned kl = lds_off32(&ktile[buf][0][0]);
#pragma unroll
    for (int i = 0; i < 4; ++i) {
      const int e = (tid + i * 128) * 8;          // bf16 elements, 16B chunks
      async_ld_b128(kl + e * 2, kg + e);
    }
    // V tile: 64 hd-rows, each 64 keys contiguous, row stride T_
    const unsigned vl = lds_off32(&vtile[buf][0][0]);
#pragma unroll
    for (int i = 0; i < 4; ++i) {
      const int c = tid + i * 128;                // 0..511 chunks
      const int row = c >> 3, sub = c & 7;
      async_ld_b128(vl + (row * 64 + sub * 8) * 2,
                    vb + (size_t)row * T_ + kblk + sub * 8);
    }
  };

  // Q A-fragments (pre-scaled by 1/sqrt(HD) at projection time)
  U16 aq[2];
#pragma unroll
  for (int ks = 0; ks < 2; ++ks) {
    const __bf16* ap = qb + (size_t)(q0 + col) * HD_ + ks * 32 + half * 16;
    aq[ks].h8[0] = *(const v8bf*)(ap);
    aq[ks].h8[1] = *(const v8bf*)(ap + 8);
  }

  stage(0, 0);
  wait_async0();
  __syncthreads();

  v8f o[4] = {};
  float mrow[8], lrow[8];
#pragma unroll
  for (int r = 0; r < 8; ++r) { mrow[r] = -3.0e38f; lrow[r] = 0.0f; }

  for (int k0 = 0, it = 0; k0 < T_; k0 += 64, ++it) {
    const int buf = it & 1;
    if (k0 + 64 < T_) stage(buf ^ 1, k0 + 64);   // overlap DMA with compute

    // ---- S tile 16x64 = Q(16x64) @ K^T(64x64): 8 WMMAs, B-frags from LDS --
    v8f s[4] = {};
#pragma unroll
    for (int nt = 0; nt < 4; ++nt) {
#pragma unroll
      for (int ks = 0; ks < 2; ++ks) {
        v16bf bf_ = *(const v16bf*)&ktile[buf][nt * 16 + col][ks * 32 + half * 16];
        s[nt] = wmma_bf16(aq[ks].v, bf_, s[nt]);
      }
    }

    // ---- online softmax, in place on s, stage-batched swizzle reductions --
    float vmax[8];
#pragma unroll
    for (int r = 0; r < 8; ++r)
      vmax[r] = fmaxf(fmaxf(s[0][r], s[1][r]), fmaxf(s[2][r], s[3][r]));
#pragma unroll
    for (int r = 0; r < 8; ++r) vmax[r] = fmaxf(vmax[r], swz<1>(vmax[r]));
#pragma unroll
    for (int r = 0; r < 8; ++r) vmax[r] = fmaxf(vmax[r], swz<2>(vmax[r]));
#pragma unroll
    for (int r = 0; r < 8; ++r) vmax[r] = fmaxf(vmax[r], swz<4>(vmax[r]));
#pragma unroll
    for (int r = 0; r < 8; ++r) vmax[r] = fmaxf(vmax[r], swz<8>(vmax[r]));

    float psum[8];
#pragma unroll
    for (int r = 0; r < 8; ++r) {
      const float mnew = fmaxf(mrow[r], vmax[r]);
      float ps = 0.0f;
#pragma unroll
      for (int nt = 0; nt < 4; ++nt) {
        const float p = __expf(s[nt][r] - mnew);
        ps += p;
        plds[wave][half * 8 + r][nt * 16 + col] = (__bf16)p;
      }
      psum[r] = ps;
      const float alpha = __expf(mrow[r] - mnew);
      mrow[r] = mnew;
      lrow[r] *= alpha;
#pragma unroll
      for (int nt2 = 0; nt2 < 4; ++nt2) o[nt2][r] *= alpha;
    }
#pragma unroll
    for (int r = 0; r < 8; ++r) psum[r] += swz<1>(psum[r]);
#pragma unroll
    for (int r = 0; r < 8; ++r) psum[r] += swz<2>(psum[r]);
#pragma unroll
    for (int r = 0; r < 8; ++r) psum[r] += swz<4>(psum[r]);
#pragma unroll
    for (int r = 0; r < 8; ++r) psum[r] += swz<8>(psum[r]);
#pragma unroll
    for (int r = 0; r < 8; ++r) lrow[r] += psum[r];

    __asm volatile("s_wait_dscnt 0x0" ::: "memory");
    // ---- P A-fragments (C-layout -> A-layout via LDS) ---------------------
    U16 apf[2];
#pragma unroll
    for (int ks = 0; ks < 2; ++ks) {
      const __bf16* pp = &plds[wave][col][ks * 32 + half * 16];
      apf[ks].h8[0] = *(const v8bf*)(pp);
      apf[ks].h8[1] = *(const v8bf*)(pp + 8);
    }
    // ---- O += P(16x64) @ V(64x64): 8 WMMAs, B-frags from LDS --------------
#pragma unroll
    for (int nt2 = 0; nt2 < 4; ++nt2) {
#pragma unroll
      for (int ks = 0; ks < 2; ++ks) {
        v16bf bf_ = *(const v16bf*)&vtile[buf][nt2 * 16 + col][ks * 32 + half * 16];
        o[nt2] = wmma_bf16(apf[ks].v, bf_, o[nt2]);
      }
    }

    wait_async0();      // next buffer landed
    __syncthreads();    // all waves done reading current buffer
  }

  // normalize and store to (B,T,D) bf16
#pragma unroll
  for (int r = 0; r < 8; ++r) {
    const float inv = 1.0f / lrow[r];
    const int t = q0 + half * 8 + r;
#pragma unroll
    for (int nt2 = 0; nt2 < 4; ++nt2) {
      const int c = h * HD_ + nt2 * 16 + col;
      ao[(size_t)(b * T_ + t) * D_ + c] = (__bf16)(o[nt2][r] * inv);
    }
  }
}

// ---------------------------------------------------------------------------
extern "C" void kernel_launch(void* const* d_in, const int* in_sizes, int n_in,
                              void* d_out, int out_size, void* d_ws, size_t ws_size,
                              hipStream_t stream) {
  const float* x  = (const float*)d_in[0];
  const float* Wq = (const float*)d_in[1];
  const float* bq = (const float*)d_in[2];
  const float* Wk = (const float*)d_in[3];
  const float* bk = (const float*)d_in[4];
  const float* Wv = (const float*)d_in[5];
  const float* bv = (const float*)d_in[6];
  const float* Wo = (const float*)d_in[7];
  const float* bo = (const float*)d_in[8];
  float* out = (float*)d_out;

  size_t off = 0;
  auto alloc = [&](size_t bytes) -> void* {
    void* p = (char*)d_ws + off;
    off += (bytes + 255) & ~(size_t)255;
    return p;
  };
  __bf16* xb  = (__bf16*)alloc((size_t)M_ * D_ * 2);
  __bf16* Wqb = (__bf16*)alloc((size_t)D_ * D_ * 2);
  __bf16* Wkb = (__bf16*)alloc((size_t)D_ * D_ * 2);
  __bf16* Wvb = (__bf16*)alloc((size_t)D_ * D_ * 2);
  __bf16* Wob = (__bf16*)alloc((size_t)D_ * D_ * 2);
  __bf16* qr  = (__bf16*)alloc((size_t)M_ * D_ * 2);   // (B,H,T,HD), pre-scaled
  __bf16* kr  = (__bf16*)alloc((size_t)M_ * D_ * 2);   // (B,H,T,HD)
  __bf16* vt  = (__bf16*)alloc((size_t)M_ * D_ * 2);   // (B,H,HD,T)
  __bf16* ao  = (__bf16*)alloc((size_t)M_ * D_ * 2);   // (B,T,D)

  // 1) fp32 -> bf16 conversions
  cvt_f32_bf16<<<2048, 256, 0, stream>>>(x,  xb,  M_ * D_);
  cvt_f32_bf16<<<512,  256, 0, stream>>>(Wq, Wqb, D_ * D_);
  cvt_f32_bf16<<<512,  256, 0, stream>>>(Wk, Wkb, D_ * D_);
  cvt_f32_bf16<<<512,  256, 0, stream>>>(Wv, Wvb, D_ * D_);
  cvt_f32_bf16<<<512,  256, 0, stream>>>(Wo, Wob, D_ * D_);

  // 2) QKV projections; RoPE fused for Q/K; 1/sqrt(HD) folded into Q
  dim3 ggrid(D_ / 64, M_ / 64);   // (16, 128)
  dim3 gblk(128);
  gemm_wmma<0><<<ggrid, gblk, 0, stream>>>(xb, Wqb, bq, qr, M_, D_, D_, 0.125f);
  gemm_wmma<0><<<ggrid, gblk, 0, stream>>>(xb, Wkb, bk, kr, M_, D_, D_, 1.0f);
  gemm_wmma<1><<<ggrid, gblk, 0, stream>>>(xb, Wvb, bv, vt, M_, D_, D_, 1.0f);

  // 3) Flash attention (64-key blocks, async double-buffered K/V)
  dim3 agrid(T_ / 64, B_ * H_);   // (32, 64)
  flash_attn<<<agrid, dim3(128), 0, stream>>>(qr, kr, vt, ao);

  // 4) Output projection, fp32 epilogue + bias
  gemm_wmma<2><<<ggrid, gblk, 0, stream>>>(ao, Wob, bo, out, M_, D_, D_, 1.0f);
}
